// MLAAttention_4999341932885
// MI455X (gfx1250) — matrix operand order repaired and checked by reading, not compiled, not run on hardware
//
#include <hip/hip_runtime.h>

// ---------------------------------------------------------------------------
// MLA attention prefill for gfx1250 (MI455X), bf16 WMMA everywhere.
// ---------------------------------------------------------------------------

constexpr int Tn   = 2048;
constexpr int HID  = 5120;
constexpr int NH   = 16;
constexpr int DN   = 128;   // d_nope
constexpr int DR   = 64;    // d_rope
constexpr int DV   = 128;   // d_v
constexpr int QL   = 1536;  // q lora
constexpr int KVL  = 512;   // kv lora
constexpr int QKH  = DN + DR;          // 192
constexpr float MLA_SCALE = 0.07216878364870322f;  // 192^-0.5

typedef __bf16 bf16x16 __attribute__((ext_vector_type(16)));
typedef float  f32x8   __attribute__((ext_vector_type(8)));

union Frag {
    bf16x16 v;
    unsigned int u[8];
};

static __device__ __forceinline__ unsigned short f2bf(float f) {
    unsigned int u = __float_as_uint(f);
    unsigned int r = (u + 0x7FFFu + ((u >> 16) & 1u)) >> 16;   // RNE
    return (unsigned short)r;
}

// --------------------------- fp32 -> bf16 convert ---------------------------
__global__ void k_f32_to_bf16(const float* __restrict__ in,
                              unsigned short* __restrict__ out, size_t n) {
    for (size_t i = (size_t)blockIdx.x * blockDim.x + threadIdx.x; i < n;
         i += (size_t)gridDim.x * blockDim.x)
        out[i] = f2bf(in[i]);
}

// ------------------------------- RMSNorm ------------------------------------
// one block per row; f32 in (row stride rsIn, length L) -> bf16 out (stride rsOut)
__global__ __launch_bounds__(256) void k_rmsnorm(
    const float* __restrict__ x, const float* __restrict__ w,
    unsigned short* __restrict__ out, int L, int rsIn, int rsOut) {
    __shared__ float red[8];
    int t = blockIdx.x, tid = threadIdx.x;
    const float* xr = x + (size_t)t * rsIn;
    float s = 0.f;
    for (int i = tid; i < L; i += 256) { float v = xr[i]; s += v * v; }
#pragma unroll
    for (int off = 16; off >= 1; off >>= 1) s += __shfl_xor(s, off, 32);
    if ((tid & 31) == 0) red[tid >> 5] = s;
    __syncthreads();
    if (tid == 0) {
        float tot = 0.f;
#pragma unroll
        for (int i = 0; i < 8; ++i) tot += red[i];
        red[0] = tot;
    }
    __syncthreads();
    float sc = rsqrtf(red[0] / (float)L + 1e-6f);
    unsigned short* o = out + (size_t)t * rsOut;
    for (int i = tid; i < L; i += 256) o[i] = f2bf(xr[i] * sc * w[i]);
}

// ------------------- generic bf16 GEMM: C = A(MxK) * B(KxN) -----------------
// Block: 128 threads (4 waves), 128x64 tile; each wave 32x64 via 2x4 WMMA.
__global__ __launch_bounds__(128) void k_gemm_bf16(
    const unsigned short* __restrict__ A,  // M x K, row major
    const unsigned short* __restrict__ B,  // K x N, row major
    float* __restrict__ C,                 // M x N, row major
    int M, int N, int K) {
    __shared__ unsigned short lA[128 * 32];  // [m][k]
    __shared__ unsigned short lBT[64 * 32];  // [n][k] (transposed)
    const int tid  = threadIdx.x;
    const int wave = tid >> 5;
    const int lane = tid & 31;
    const int l16  = lane & 15;
    const int lh   = lane >> 4;
    const int bm = blockIdx.y * 128;
    const int bn = blockIdx.x * 64;
    const int wm = wave * 32;

    f32x8 acc[2][4] = {};

    const int bk  = tid >> 2;               // 0..31
    const int bn0 = (tid & 3) * 16;         // 0/16/32/48

    for (int k0 = 0; k0 < K; k0 += 32) {
        // stage A tile (128x32): one 64B row per thread
        {
            const uint4* src = (const uint4*)(A + (size_t)(bm + tid) * K + k0);
            uint4 d0 = src[0], d1 = src[1], d2 = src[2], d3 = src[3];
            uint4* dst = (uint4*)(lA + tid * 32);
            dst[0] = d0; dst[1] = d1; dst[2] = d2; dst[3] = d3;
        }
        // stage B tile transposed -> lBT[n][k]
        {
            const unsigned short* src = B + (size_t)(k0 + bk) * N + bn + bn0;
            uint4 d0 = ((const uint4*)src)[0];
            uint4 d1 = ((const uint4*)src)[1];
            unsigned short tmp[16];
            *((uint4*)tmp) = d0; *((uint4*)(tmp + 8)) = d1;
#pragma unroll
            for (int i = 0; i < 16; ++i) lBT[(bn0 + i) * 32 + bk] = tmp[i];
        }
        if (k0 + 32 < K) {
            __builtin_prefetch(A + (size_t)(bm + tid) * K + k0 + 32, 0, 3);
            __builtin_prefetch(B + (size_t)(k0 + 32 + bk) * N + bn + bn0, 0, 3);
        }
        __syncthreads();

        Frag a[2], b[4];
#pragma unroll
        for (int i = 0; i < 2; ++i) {
            const unsigned short* abase = lA + (wm + i * 16 + l16) * 32;
#pragma unroll
            for (int v = 0; v < 8; ++v) {
                int ka = lh * 16 + v * 2;
                a[i].u[v] = *(const unsigned int*)(abase + ka);
            }
        }
#pragma unroll
        for (int j = 0; j < 4; ++j) {
            const unsigned short* bbase = lBT + (j * 16 + l16) * 32;
#pragma unroll
            for (int v = 0; v < 8; ++v)
                b[j].u[v] = *(const unsigned int*)(bbase + lh * 16 + v * 2);
        }
#pragma unroll
        for (int i = 0; i < 2; ++i)
#pragma unroll
            for (int j = 0; j < 4; ++j)
                acc[i][j] = __builtin_amdgcn_wmma_f32_16x16x32_bf16(
                    false, a[i].v, false, b[j].v, (short)0, acc[i][j], false, false);
        __syncthreads();
    }
#pragma unroll
    for (int i = 0; i < 2; ++i)
#pragma unroll
        for (int j = 0; j < 4; ++j)
#pragma unroll
            for (int v = 0; v < 8; ++v) {
                int m = bm + wm + i * 16 + v + 8 * lh;
                int n = bn + j * 16 + l16;
                C[(size_t)m * N + n] = acc[i][j][v];
            }
}

// ---------------- RoPE + pack q (f32 [T,NH*192] -> bf16) --------------------
__global__ __launch_bounds__(64) void k_rope_q(
    const float* __restrict__ q, const int* __restrict__ pos,
    unsigned short* __restrict__ qb) {
    int t = blockIdx.x, h = blockIdx.y, tid = threadIdx.x;
    const float* src = q + (size_t)t * (NH * QKH) + h * QKH;
    unsigned short* dst = qb + (size_t)t * (NH * QKH) + h * QKH;
    dst[tid]      = f2bf(src[tid]);
    dst[tid + 64] = f2bf(src[tid + 64]);
    if (tid < 32) {
        float inv = __powf(10000.f, -(float)(2 * tid) / 64.f);
        float ang = (float)pos[t] * inv;
        float cs = __cosf(ang), sn = __sinf(ang);
        float xe = src[DN + 2 * tid], xo = src[DN + 2 * tid + 1];
        dst[DN + 2 * tid]     = f2bf(xe * cs - xo * sn);
        dst[DN + 2 * tid + 1] = f2bf(xo * cs + xe * sn);
    }
}

// ------- build k_full (nope||roped pe) and transposed V [h*128+d][t] --------
__global__ __launch_bounds__(64) void k_build_kv(
    const float* __restrict__ kvup,   // [T, NH*256]
    const float* __restrict__ kv,     // [T, 576] (cols 512..575 = k_pe)
    const int* __restrict__ pos,
    unsigned short* __restrict__ kf,  // [T, NH*192]
    unsigned short* __restrict__ vT)  // [NH*128, T]
{
    int t = blockIdx.x, h = blockIdx.y, tid = threadIdx.x;
    const float* u = kvup + (size_t)t * (NH * (DN + DV)) + h * (DN + DV);
    unsigned short* kd = kf + (size_t)t * (NH * QKH) + h * QKH;
    kd[tid]      = f2bf(u[tid]);
    kd[tid + 64] = f2bf(u[tid + 64]);
    vT[(size_t)(h * DV + tid) * Tn + t]      = f2bf(u[DN + tid]);
    vT[(size_t)(h * DV + 64 + tid) * Tn + t] = f2bf(u[DN + 64 + tid]);
    if (tid < 32) {
        float inv = __powf(10000.f, -(float)(2 * tid) / 64.f);
        float ang = (float)pos[t] * inv;
        float cs = __cosf(ang), sn = __sinf(ang);
        const float* pe = kv + (size_t)t * (KVL + DR) + KVL;
        float xe = pe[2 * tid], xo = pe[2 * tid + 1];
        kd[DN + 2 * tid]     = f2bf(xe * cs - xo * sn);
        kd[DN + 2 * tid + 1] = f2bf(xo * cs + xe * sn);
    }
}

// -------------------- flash attention, 1 wave per (h, 16-q tile) ------------
__global__ __launch_bounds__(32) void k_mla_attn(
    const unsigned short* __restrict__ qb,  // [T, NH*192] bf16
    const unsigned short* __restrict__ kf,  // [T, NH*192] bf16
    const unsigned short* __restrict__ vT,  // [NH*128, T] bf16
    unsigned short* __restrict__ attnb)     // [T, NH*128] bf16
{
    __shared__ unsigned short lP[16 * 32];
    const int lane = threadIdx.x & 31;
    const int l16 = lane & 15, lh = lane >> 4;
    const int h  = blockIdx.y;
    const int m0 = blockIdx.x * 16;

    // Q fragments: 6 K-chunks of 32 over head dim 192
    Frag aq[6];
#pragma unroll
    for (int c = 0; c < 6; ++c) {
        const unsigned short* base =
            qb + (size_t)(m0 + l16) * (NH * QKH) + h * QKH + c * 32;
#pragma unroll
        for (int v = 0; v < 8; ++v) {
            int ka = lh * 16 + v * 2;
            aq[c].u[v] = *(const unsigned int*)(base + ka);
        }
    }

    f32x8 acc[8] = {};
    float mstate[8], lstate[8];
#pragma unroll
    for (int v = 0; v < 8; ++v) { mstate[v] = -1e30f; lstate[v] = 0.f; }

    for (int s0 = 0; s0 <= m0 + 15; s0 += 32) {
        // scores for two 16-key halves
        f32x8 S[2];
#pragma unroll
        for (int sub = 0; sub < 2; ++sub) {
            f32x8 c = {};
            const unsigned short* kb =
                kf + (size_t)(s0 + sub * 16 + l16) * (NH * QKH) + h * QKH;
#pragma unroll
            for (int ch = 0; ch < 6; ++ch) {
                Frag b;
#pragma unroll
                for (int v = 0; v < 8; ++v)
                    b.u[v] = *(const unsigned int*)(kb + ch * 32 + lh * 16 + v * 2);
                c = __builtin_amdgcn_wmma_f32_16x16x32_bf16(
                    false, aq[ch].v, false, b.v, (short)0, c, false, false);
            }
            S[sub] = c;
        }
        // online softmax over the 32 keys
#pragma unroll
        for (int v = 0; v < 8; ++v) {
            int qi = m0 + v + 8 * lh;
            float s0v = S[0][v] * MLA_SCALE;
            float s1v = S[1][v] * MLA_SCALE;
            if (s0 + l16 > qi)      s0v = -1e30f;
            if (s0 + 16 + l16 > qi) s1v = -1e30f;
            float r = fmaxf(s0v, s1v);
#pragma unroll
            for (int off = 8; off >= 1; off >>= 1)
                r = fmaxf(r, __shfl_xor(r, off, 16));
            float nm = fmaxf(mstate[v], r);
            float sc = __expf(mstate[v] - nm);
            float p0 = __expf(s0v - nm);
            float p1 = __expf(s1v - nm);
            float rs = p0 + p1;
#pragma unroll
            for (int off = 8; off >= 1; off >>= 1)
                rs += __shfl_xor(rs, off, 16);
            lstate[v] = lstate[v] * sc + rs;
            mstate[v] = nm;
#pragma unroll
            for (int c2 = 0; c2 < 8; ++c2) acc[c2][v] *= sc;
            int row = v + 8 * lh;
            lP[row * 32 + l16]      = f2bf(p0);
            lP[row * 32 + 16 + l16] = f2bf(p1);
        }
        __syncthreads();
        // reload P in A-fragment layout (16x32)
        Frag ap;
#pragma unroll
        for (int v = 0; v < 8; ++v) {
            int ka = lh * 16 + v * 2;
            ap.u[v] = *(const unsigned int*)(lP + l16 * 32 + ka);
        }
        // acc += P @ V over 8 output chunks of 16
#pragma unroll
        for (int c = 0; c < 8; ++c) {
            Frag bv;
            const unsigned short* vbase =
                vT + (size_t)(h * DV + c * 16 + l16) * Tn + s0;
#pragma unroll
            for (int v = 0; v < 8; ++v)
                bv.u[v] = *(const unsigned int*)(vbase + lh * 16 + v * 2);
            acc[c] = __builtin_amdgcn_wmma_f32_16x16x32_bf16(
                false, ap.v, false, bv.v, (short)0, acc[c], false, false);
        }
        __syncthreads();
    }
    // epilogue: normalize and store bf16
#pragma unroll
    for (int v = 0; v < 8; ++v) {
        float inv = 1.0f / lstate[v];
        int t = m0 + v + 8 * lh;
#pragma unroll
        for (int c = 0; c < 8; ++c)
            attnb[(size_t)t * (NH * DV) + h * DV + c * 16 + l16] =
                f2bf(acc[c][v] * inv);
    }
}

// ---------------------------------------------------------------------------
extern "C" void kernel_launch(void* const* d_in, const int* in_sizes, int n_in,
                              void* d_out, int out_size, void* d_ws, size_t ws_size,
                              hipStream_t stream) {
    const int*   positions = (const int*)d_in[0];
    const float* hidden    = (const float*)d_in[1];
    const float* w_qa      = (const float*)d_in[2];
    const float* qa_ln_w   = (const float*)d_in[3];
    const float* w_qb      = (const float*)d_in[4];
    const float* w_kva     = (const float*)d_in[5];
    const float* kva_ln_w  = (const float*)d_in[6];
    const float* w_kvb     = (const float*)d_in[7];
    const float* w_o       = (const float*)d_in[8];
    float* out = (float*)d_out;

    char* base = (char*)d_ws;
    size_t off = 0;
    auto alloc = [&](size_t bytes) -> void* {
        void* p = base + off;
        off += (bytes + 255) & ~(size_t)255;
        return p;
    };
    unsigned short* hsb   = (unsigned short*)alloc((size_t)Tn * HID * 2);
    unsigned short* Wb    = (unsigned short*)alloc((size_t)(NH * DV) * HID * 2); // 21MB, max weight
    float*          bigF  = (float*)alloc((size_t)Tn * 4096 * 4);                // q_c / q / kv_up
    unsigned short* qcb   = (unsigned short*)alloc((size_t)Tn * QL * 2);
    unsigned short* qb    = (unsigned short*)alloc((size_t)Tn * NH * QKH * 2);
    float*          kvf   = (float*)alloc((size_t)Tn * (KVL + DR) * 4);
    unsigned short* kvcb  = (unsigned short*)alloc((size_t)Tn * KVL * 2);
    unsigned short* kfb   = (unsigned short*)alloc((size_t)Tn * NH * QKH * 2);
    unsigned short* vTb   = (unsigned short*)alloc((size_t)NH * DV * Tn * 2);
    unsigned short* attnb = (unsigned short*)alloc((size_t)Tn * NH * DV * 2);

    auto conv = [&](const float* src, unsigned short* dst, size_t n) {
        k_f32_to_bf16<<<2048, 256, 0, stream>>>(src, dst, n);
    };
    auto gemm = [&](const unsigned short* A, const unsigned short* B, float* C,
                    int M, int N, int K) {
        k_gemm_bf16<<<dim3(N / 64, M / 128), 128, 0, stream>>>(A, B, C, M, N, K);
    };

    // 1) hidden -> bf16
    conv(hidden, hsb, (size_t)Tn * HID);

    // 2) q_c = hidden @ w_qa ; rmsnorm
    conv(w_qa, Wb, (size_t)HID * QL);
    gemm(hsb, Wb, bigF, Tn, QL, HID);
    k_rmsnorm<<<Tn, 256, 0, stream>>>(bigF, qa_ln_w, qcb, QL, QL, QL);

    // 3) q = q_c @ w_qb ; rope + pack bf16
    conv(w_qb, Wb, (size_t)QL * NH * QKH);
    gemm(qcb, Wb, bigF, Tn, NH * QKH, QL);
    k_rope_q<<<dim3(Tn, NH), 64, 0, stream>>>(bigF, positions, qb);

    // 4) kv = hidden @ w_kva ; rmsnorm of first 512 cols
    conv(w_kva, Wb, (size_t)HID * (KVL + DR));
    gemm(hsb, Wb, kvf, Tn, KVL + DR, HID);
    k_rmsnorm<<<Tn, 256, 0, stream>>>(kvf, kva_ln_w, kvcb, KVL, KVL + DR, KVL);

    // 5) kv_up = kv_c @ w_kvb ; build k_full + transposed V
    conv(w_kvb, Wb, (size_t)KVL * NH * (DN + DV));
    gemm(kvcb, Wb, bigF, Tn, NH * (DN + DV), KVL);
    k_build_kv<<<dim3(Tn, NH), 64, 0, stream>>>(bigF, kvf, positions, kfb, vTb);

    // 6) attention
    k_mla_attn<<<dim3(Tn / 16, NH), 32, 0, stream>>>(qb, kfb, vTb, attnb);

    // 7) out = attn @ w_o
    conv(w_o, Wb, (size_t)(NH * DV) * HID);
    gemm(attnb, Wb, out, Tn, HID, NH * DV);
}
